// Generator_83399674953737
// MI455X (gfx1250) — matrix operand order repaired and checked by reading, not compiled, not run on hardware
//
#include <hip/hip_runtime.h>
#include <cstdint>
#include <cstddef>

typedef __attribute__((ext_vector_type(16))) _Float16 v16h;
typedef __attribute__((ext_vector_type(8)))  float    v8f;
typedef __attribute__((ext_vector_type(4)))  uint32_t v4u;
typedef _Float16 h16;

// ---------------------------------------------------------------------------
// Layout converters / weight packing (32-bit indexing throughout)
// ---------------------------------------------------------------------------
__global__ void k_nchw_f32_to_nhwc_f16(const float* __restrict__ src, h16* __restrict__ dst,
                                       int B, int C, int H, int W) {
  int idx = blockIdx.x * blockDim.x + threadIdx.x;
  int total = B * C * H * W;
  if (idx >= total) return;
  int c = idx % C;
  int r = idx / C;
  int x = r % W; r /= W;
  int y = r % H;
  int b = r / H;
  dst[idx] = (h16)src[((b * C + c) * H + y) * W + x];
}

// torch w layout [Cin][Cout][3][3] -> wf [9][CoutPad][Cin] (f16), tap t = i*3+j
__global__ void k_pack_weights(const float* __restrict__ w, h16* __restrict__ wf,
                               int Cin, int Cout, int CoutPad) {
  int idx = blockIdx.x * blockDim.x + threadIdx.x;
  int total = 9 * CoutPad * Cin;
  if (idx >= total) return;
  int ci = idx % Cin;
  int co = (idx / Cin) % CoutPad;
  int t  = idx / (Cin * CoutPad);
  int ki = t / 3, kj = t % 3;
  float v = (co < Cout) ? w[((ci * Cout + co) * 3 + ki) * 3 + kj] : 0.0f;
  wf[idx] = (h16)v;
}

// ---------------------------------------------------------------------------
// SepConv fusion (K=5, replication pad, mask blend)
// in: NHWC f16; kv/kh: [B][5][H][W] f32; mask: [B][1][H][W] f32
// outN: f32 NCHW (optional); outH: f16 NHWC (optional)
// ---------------------------------------------------------------------------
__global__ void k_fusion(const h16* __restrict__ in, const float* __restrict__ kv,
                         const float* __restrict__ kh, const float* __restrict__ mk,
                         float* __restrict__ outN, h16* __restrict__ outH,
                         int B, int C, int H, int W) {
  int idx = blockIdx.x * blockDim.x + threadIdx.x;
  int total = B * H * W * C;
  if (idx >= total) return;
  int c = idx % C;
  int r = idx / C;
  int x = r % W; r /= W;
  int y = r % H;
  int b = r / H;

  float kvv[5], khh[5];
#pragma unroll
  for (int i = 0; i < 5; ++i) {
    kvv[i] = kv[(((b * 5) + i) * H + y) * W + x];
    khh[i] = kh[(((b * 5) + i) * H + y) * W + x];
  }
  float acc = 0.0f;
#pragma unroll
  for (int i = 0; i < 5; ++i) {
    int yy = y + i - 2; yy = yy < 0 ? 0 : (yy >= H ? H - 1 : yy);
    float rowacc = 0.0f;
#pragma unroll
    for (int j = 0; j < 5; ++j) {
      int xx = x + j - 2; xx = xx < 0 ? 0 : (xx >= W ? W - 1 : xx);
      rowacc += (float)in[((b * H + yy) * W + xx) * C + c] * khh[j];
    }
    acc += kvv[i] * rowacc;
  }
  float mv  = mk[(b * H + y) * W + x];
  float inc = (float)in[((b * H + y) * W + x) * C + c];
  float res = mv * acc + (1.0f - mv) * inc;
  if (outN) outN[((b * C + c) * H + y) * W + x] = res;
  if (outH) outH[idx] = (h16)res;
}

// ---------------------------------------------------------------------------
// Implicit-GEMM transposed conv, k=3, p=1 (S=1) or stride 2 + output_padding 1
// (S=2). One wave -> 16(co) x 16(pixel) tile via v_wmma_f32_16x16x32_f16.
// Grid: x = pixel tiles (8 waves/block), y = co tile, z = batch. Wout, Hout*Wout
// are powers of two -> pure shift/mask addressing, zero integer division.
// Two K-interleaved accumulators break the WMMA->WMMA C-chain so independent
// wmma ops can issue back-to-back (ISA 7.12.1 hazard table).
// act: 0 = relu, 1 = tanh
// ---------------------------------------------------------------------------
template <int S>
__global__ __launch_bounds__(256) void k_deconv_wmma(
    const h16* __restrict__ in, const h16* __restrict__ wf, h16* __restrict__ out,
    int Cin, int Hin, int Win, int Cout, int CoutPad, int logWout, int act) {
  const int lane = threadIdx.x & 31;
  const int wave = threadIdx.x >> 5;

  const int Wout = 1 << logWout;
  const int Hout = Hin * S;                 // (Hout*Wout) = hw, power of two
  const int nPixTiles = (Hout * Wout) >> 4;
  const int pixTile = blockIdx.x * 8 + wave;      // wave-uniform
  if (pixTile >= nPixTiles) return;
  const int coT = blockIdx.y;
  const int b   = blockIdx.z;

  // lane geometry (16-bit WMMA fragments, wave32; contiguous K per lane half, not ISA table order)
  const int n        = lane & 15;   // B/C/D column; also A row (m == n)
  const int khalfSel = lane >> 4;   // 0: K 0-15 (A and B); 1: K 16-31 (A and B)

  const int rem = pixTile * 16 + n;         // flattened y*Wout+x within image
  const int y   = rem >> logWout;
  const int x   = rem & (Wout - 1);
  const int co  = coT * 16 + n;

  v8f acc0 = {};
  v8f acc1 = {};

  for (int t = 0; t < 9; ++t) {
    const int ki = t / 3, kj = t % 3;       // compile-time unrollable
    const int ny = y + 1 - ki;
    const int nx = x + 1 - kj;
    int iy, ix;
    bool valid;
    if (S == 2) {
      valid = ((ny | nx) >= 0) && (((ny & 1) | (nx & 1)) == 0);
      iy = ny >> 1; ix = nx >> 1;
      valid = valid && (iy < Hin) && (ix < Win);
    } else {
      iy = ny; ix = nx;
      valid = ((unsigned)iy < (unsigned)Hin) && ((unsigned)ix < (unsigned)Win);
    }

    // A row: wf[t][co][0..Cin); B row: in[b][iy][ix][0..Cin)  (16B aligned)
    const v4u* __restrict__ aq =
        (const v4u*)(wf + (t * CoutPad + co) * Cin);
    const v4u* __restrict__ bq = valid
        ? (const v4u*)(in + ((b * Hin + iy) * Win + ix) * Cin) + khalfSel * 2
        : nullptr;

    for (int cib = 0; cib < Cin; cib += 64) {
      union { v16h v; v4u q[2]; } A0, B0, A1, B1;
      const int aqb0 = cib >> 3;            // v4u index of 32-ch chunk 0
      const int aqb1 = (cib + 32) >> 3;
      // A (16x32 f16): lanes<16 take K 0-15 (q0,q1); lanes>=16: K 16-31
      A0.q[0] = aq[aqb0 + 2 * khalfSel];
      A0.q[1] = aq[aqb0 + 2 * khalfSel + 1];
      A1.q[0] = aq[aqb1 + 2 * khalfSel];
      A1.q[1] = aq[aqb1 + 2 * khalfSel + 1];
      // B (32x16 f16): lane half selects K 0-15 / 16-31, 2 channels per dword
      if (valid) {
        B0.q[0] = bq[aqb0];
        B0.q[1] = bq[aqb0 + 1];
        B1.q[0] = bq[aqb1];
        B1.q[1] = bq[aqb1 + 1];
      } else {
        v4u z = {};
        B0.q[0] = z; B0.q[1] = z; B1.q[0] = z; B1.q[1] = z;
      }
      acc0 = __builtin_amdgcn_wmma_f32_16x16x32_f16(
          false, A0.v, false, B0.v, (short)0, acc0, false, false);
      acc1 = __builtin_amdgcn_wmma_f32_16x16x32_f16(
          false, A1.v, false, B1.v, (short)0, acc1, false, false);
    }
  }

  // D (16x16 f32): N = lane&15, VGPR r -> M = r + 8*(lane>=16)
  h16* __restrict__ orow = out + (((b * Hout + y) << logWout) + x) * Cout;
#pragma unroll
  for (int rI = 0; rI < 8; ++rI) {
    int coOut = coT * 16 + rI + 8 * khalfSel;
    if (coOut < Cout) {
      float v = acc0[rI] + acc1[rI];
      v = (act == 0) ? (v > 0.0f ? v : 0.0f) : tanhf(v);
      orow[coOut] = (h16)v;
    }
  }
}

// ---------------------------------------------------------------------------
// Host-side orchestration
// ---------------------------------------------------------------------------
static inline void launch_deconv(int S, const h16* in, const h16* wf, h16* outp,
                                 int B, int Cin, int Hin, int Win,
                                 int Cout, int CoutPad, int act, hipStream_t s) {
  const int Hout = Hin * S, Wout = Win * S;
  int logWout = 0; while ((1 << logWout) != Wout) ++logWout;
  const int nPixTiles = (Hout * Wout) / 16;
  dim3 grid((nPixTiles + 7) / 8, CoutPad / 16, B);
  if (S == 1)
    k_deconv_wmma<1><<<grid, 256, 0, s>>>(in, wf, outp, Cin, Hin, Win, Cout, CoutPad, logWout, act);
  else
    k_deconv_wmma<2><<<grid, 256, 0, s>>>(in, wf, outp, Cin, Hin, Win, Cout, CoutPad, logWout, act);
}

static inline void launch_fusion(const h16* in, const float* kv, const float* kh,
                                 const float* mk, float* outN, h16* outH,
                                 int B, int C, int H, int W, hipStream_t s) {
  int total = B * H * W * C;
  int blocks = (total + 255) / 256;
  k_fusion<<<blocks, 256, 0, s>>>(in, kv, kh, mk, outN, outH, B, C, H, W);
}

extern "C" void kernel_launch(void* const* d_in, const int* in_sizes, int n_in,
                              void* d_out, int out_size, void* d_ws, size_t ws_size,
                              hipStream_t stream) {
  (void)in_sizes; (void)n_in; (void)out_size; (void)ws_size;
  const int B = 128;

  const float* cont = (const float*)d_in[0];
  const float* kv[4] = {(const float*)d_in[1],  (const float*)d_in[4],
                        (const float*)d_in[7],  (const float*)d_in[10]};
  const float* kh[4] = {(const float*)d_in[2],  (const float*)d_in[5],
                        (const float*)d_in[8],  (const float*)d_in[11]};
  const float* mk[4] = {(const float*)d_in[3],  (const float*)d_in[6],
                        (const float*)d_in[9],  (const float*)d_in[12]};
  const float* w[6]  = {(const float*)d_in[13], (const float*)d_in[14],
                        (const float*)d_in[15], (const float*)d_in[16],
                        (const float*)d_in[17], (const float*)d_in[18]};

  // ---- workspace layout (bump allocator, region reuse; ~140 MB total) ----
  char* ws = (char*)d_ws;
  size_t off = 0;
  auto alloc = [&](size_t bytes) { size_t o = off; off += (bytes + 255) & ~(size_t)255; return o; };
  const size_t R0  = alloc(33554432);                 // t16a -> h16 -> t32b -> o64
  const size_t R1  = alloc(33554432);                 // t16b -> t32a -> h32
  const size_t R2  = alloc(67108864);                 // h64
  const size_t X8A = alloc((size_t)B * 64 * 8 * 8 * 2);
  const size_t X8B = alloc((size_t)B * 64 * 8 * 8 * 2);
  const int cinA[6]  = {64, 512, 256, 128, 128, 64};
  const int coutA[6] = {512, 256, 128, 128, 64, 3};
  const int coutP[6] = {512, 256, 128, 128, 64, 16};
  size_t WF[6];
  for (int l = 0; l < 6; ++l) WF[l] = alloc(9ull * coutP[l] * cinA[l] * 2);

  // ---- d_out segments (return order: out, feat_8, feat_16, feat_32) ----
  float* oOUT = (float*)d_out;
  float* oF8  = oOUT + 1572864;
  float* oF16 = oF8  + 524288;
  float* oF32 = oF16 + 16777216;

  // ---- weight packing (f32 torch layout -> f16 [tap][CoutPad][Cin]) ----
  for (int l = 0; l < 6; ++l) {
    int n = 9 * coutP[l] * cinA[l];
    k_pack_weights<<<(n + 255) / 256, 256, 0, stream>>>(
        w[l], (h16*)(ws + WF[l]), cinA[l], coutA[l], coutP[l]);
  }

  // ---- cont_feat f32 NCHW -> f16 NHWC ----
  {
    int n = B * 64 * 8 * 8;
    k_nchw_f32_to_nhwc_f16<<<(n + 255) / 256, 256, 0, stream>>>(
        cont, (h16*)(ws + X8A), B, 64, 8, 8);
  }

  // feat_8 = fusion0(cont)
  launch_fusion((h16*)(ws + X8A), kv[0], kh[0], mk[0], oF8, (h16*)(ws + X8B), B, 64, 8, 8, stream);
  // relu(deconv1 stride2): 64x8x8 -> 512x16x16
  launch_deconv(2, (h16*)(ws + X8B), (h16*)(ws + WF[0]), (h16*)(ws + R0), B, 64, 8, 8, 512, 512, 0, stream);
  // feat_16 = fusion1
  launch_fusion((h16*)(ws + R0), kv[1], kh[1], mk[1], oF16, (h16*)(ws + R1), B, 512, 16, 16, stream);
  // relu(deconv2 stride1): 512 -> 256 @16
  launch_deconv(1, (h16*)(ws + R1), (h16*)(ws + WF[1]), (h16*)(ws + R0), B, 512, 16, 16, 256, 256, 0, stream);
  // relu(deconv3 stride2): 256x16 -> 128x32
  launch_deconv(2, (h16*)(ws + R0), (h16*)(ws + WF[2]), (h16*)(ws + R1), B, 256, 16, 16, 128, 128, 0, stream);
  // feat_32 = fusion2
  launch_fusion((h16*)(ws + R1), kv[2], kh[2], mk[2], oF32, (h16*)(ws + R0), B, 128, 32, 32, stream);
  // relu(deconv4 stride1): 128 -> 128 @32
  launch_deconv(1, (h16*)(ws + R0), (h16*)(ws + WF[3]), (h16*)(ws + R1), B, 128, 32, 32, 128, 128, 0, stream);
  // relu(deconv5 stride2): 128x32 -> 64x64
  launch_deconv(2, (h16*)(ws + R1), (h16*)(ws + WF[4]), (h16*)(ws + R2), B, 128, 32, 32, 64, 64, 0, stream);
  // tanh(deconv6 stride1): 64 -> 3 (padded to 16) @64
  launch_deconv(1, (h16*)(ws + R2), (h16*)(ws + WF[5]), (h16*)(ws + R0), B, 64, 64, 64, 3, 16, 1, stream);
  // out = fusion3
  launch_fusion((h16*)(ws + R0), kv[3], kh[3], mk[3], oOUT, nullptr, B, 3, 64, 64, stream);
}
